// XLSTMCurveForecaster_82162724373038
// MI455X (gfx1250) — matrix operand order repaired and checked by reading, not compiled, not run on hardware
//
#include <hip/hip_runtime.h>
#include <hip/hip_bf16.h>

// ---------------------------------------------------------------------------
// CDNA5 (gfx1250) xLSTM forward.  wave32, WMMA bf16 16x16x32, async LDS copy.
// Shapes (fixed): B=16 S=512 IN=32 OUT=32 EMB=512 NB=2 NH=4 K=4
//                 INNER=1024 DH=256
// ---------------------------------------------------------------------------

#define USE_ASYNC_LDS 1

typedef __attribute__((ext_vector_type(16))) __bf16 v16bf;
typedef __attribute__((ext_vector_type(8)))  float  v8f;

static __device__ __forceinline__ v8f vzero8() {
    v8f z = {0.f, 0.f, 0.f, 0.f, 0.f, 0.f, 0.f, 0.f};
    return z;
}

static __device__ __forceinline__ void bf8_from_f4x2(__bf16* dst, float4 a, float4 b) {
    dst[0] = (__bf16)a.x; dst[1] = (__bf16)a.y;
    dst[2] = (__bf16)a.z; dst[3] = (__bf16)a.w;
    dst[4] = (__bf16)b.x; dst[5] = (__bf16)b.y;
    dst[6] = (__bf16)b.z; dst[7] = (__bf16)b.w;
}

// ---------------------------------------------------------------------------
// Generic bf16 WMMA GEMM:  C[M,N] = A[M,K] * op(B)[K,N]  (+ optional residual)
// All dims must satisfy M%64==0, N%64==0, K%32==0 (true for every call here).
// Block tile 64x64, 256 threads = 8 waves, each wave owns two 16x16 D tiles.
// B tile stored transposed in LDS ([n][k]) so fragment reads are contiguous.
// ---------------------------------------------------------------------------
__global__ __launch_bounds__(256)
void k_gemm_bf16(const float* __restrict__ A, int lda, long long sA,
                 const float* __restrict__ Bm, int ldb, long long sB, int transB,
                 float* __restrict__ C, int ldc, long long sC,
                 const float* __restrict__ Res, int ldr, long long sR,
                 int M, int N, int K)
{
    __shared__ __bf16 Als[64][32];
    __shared__ __bf16 BlsT[64][32];    // [n][k]

    const int tid  = threadIdx.x;
    const int lane = tid & 31;
    const int wid  = tid >> 5;
    const int lh   = lane >> 4;   // lane half (0/1)
    const int l16  = lane & 15;

    const int m0 = blockIdx.y * 64;
    const int n0 = blockIdx.x * 64;
    const long long z = blockIdx.z;
    A  += z * sA;
    Bm += z * sB;
    C  += z * sC;
    if (Res) Res += z * sR;

    // tiles 0..15 in a 4x4 grid (r = t/4, c = t%4); wave w -> tiles w, w+8
    const int r0 = (wid)     >> 2, c0 = (wid)     & 3;
    const int r1 = (wid + 8) >> 2, c1 = (wid + 8) & 3;

    v8f acc0 = vzero8();
    v8f acc1 = vzero8();

    const int arow = tid >> 2;           // 0..63
    const int aseg = (tid & 3) * 8;      // 0,8,16,24
    // B staging (layout depends on transB, both land in BlsT[n][k])
    const int bnrow = tid >> 2;          // 0..63 (trans path: fixed n)
    const int bkseg = (tid & 3) * 8;     // along k
    const int bkrow = tid >> 3;          // 0..31 (non-trans path: fixed k)
    const int bnseg = (tid & 7) * 8;     // along n

    for (int k0 = 0; k0 < K; k0 += 32) {
        // ---- stage A 64x32 -> bf16 (vectorized, no guards) ----
        {
            const float4* ap =
                (const float4*)(A + (long long)(m0 + arow) * lda + k0 + aseg);
            float4 a0 = ap[0], a1 = ap[1];
            bf8_from_f4x2(&Als[arow][aseg], a0, a1);
        }
        // ---- stage B -> BlsT[n][k] ----
        if (transB) {
            // source contiguous along k: Bm[(n0+n)*ldb + k0+k]
            const float4* bp =
                (const float4*)(Bm + (long long)(n0 + bnrow) * ldb + k0 + bkseg);
            float4 b0 = bp[0], b1 = bp[1];
            bf8_from_f4x2(&BlsT[bnrow][bkseg], b0, b1);
        } else {
            // source contiguous along n: Bm[(k0+k)*ldb + n0+n]
            const float4* bp =
                (const float4*)(Bm + (long long)(k0 + bkrow) * ldb + n0 + bnseg);
            float4 b0 = bp[0], b1 = bp[1];
            BlsT[bnseg + 0][bkrow] = (__bf16)b0.x;
            BlsT[bnseg + 1][bkrow] = (__bf16)b0.y;
            BlsT[bnseg + 2][bkrow] = (__bf16)b0.z;
            BlsT[bnseg + 3][bkrow] = (__bf16)b0.w;
            BlsT[bnseg + 4][bkrow] = (__bf16)b1.x;
            BlsT[bnseg + 5][bkrow] = (__bf16)b1.y;
            BlsT[bnseg + 6][bkrow] = (__bf16)b1.z;
            BlsT[bnseg + 7][bkrow] = (__bf16)b1.w;
        }
        // prefetch next k tile while we compute on this one
        if (k0 + 32 < K) {
            __builtin_prefetch(A + (long long)(m0 + arow) * lda + k0 + 32 + aseg, 0, 0);
            __builtin_prefetch(Bm + (transB
                 ? ((long long)(n0 + bnrow) * ldb + k0 + 32 + bkseg)
                 : ((long long)(k0 + 32 + bkrow) * ldb + n0 + bnseg)), 0, 0);
        }
        __syncthreads();

        v16bf a0, a1, b0, b1;
        #pragma unroll
        for (int e = 0; e < 16; ++e) {
            int kk = lh * 16 + e;
            a0[e] = Als[r0 * 16 + l16][kk];
            a1[e] = Als[r1 * 16 + l16][kk];
        }
        #pragma unroll
        for (int e = 0; e < 16; ++e) {
            int kk = lh * 16 + e;                  // contiguous in BlsT[n][.]
            b0[e] = BlsT[c0 * 16 + l16][kk];
            b1[e] = BlsT[c1 * 16 + l16][kk];
        }
        acc0 = __builtin_amdgcn_wmma_f32_16x16x32_bf16(false, a0, false, b0,
                                                       (short)0, acc0, false, false);
        acc1 = __builtin_amdgcn_wmma_f32_16x16x32_bf16(false, a1, false, b1,
                                                       (short)0, acc1, false, false);
        __syncthreads();
    }

    #pragma unroll
    for (int vr = 0; vr < 8; ++vr) {
        int mrow = vr + lh * 8;
        {
            long long row = m0 + r0 * 16 + mrow, col = n0 + c0 * 16 + l16;
            float v = acc0[vr];
            if (Res) v += Res[row * ldr + col];
            C[row * ldc + col] = v;
        }
        {
            long long row = m0 + r1 * 16 + mrow, col = n0 + c1 * 16 + l16;
            float v = acc1[vr];
            if (Res) v += Res[row * ldr + col];
            C[row * ldc + col] = v;
        }
    }
}

// ---------------------------------------------------------------------------
// LayerNorm over 512 cols (no bias), y = (x-mu)*rsqrt(var+eps)*w
// ---------------------------------------------------------------------------
__global__ __launch_bounds__(256)
void k_layernorm(const float* __restrict__ x, const float* __restrict__ w,
                 float* __restrict__ y)
{
    __shared__ float r1[256], r2[256];
    const long long row = blockIdx.x;
    const int tid = threadIdx.x;
    const float* src = x + row * 512;
    float x0 = src[tid], x1 = src[tid + 256];
    r1[tid] = x0 + x1;
    r2[tid] = x0 * x0 + x1 * x1;
    __syncthreads();
    for (int off = 128; off > 0; off >>= 1) {
        if (tid < off) { r1[tid] += r1[tid + off]; r2[tid] += r2[tid + off]; }
        __syncthreads();
    }
    float mu = r1[0] * (1.f / 512.f);
    float var = r2[0] * (1.f / 512.f) - mu * mu;
    float rs = rsqrtf(var + 1e-5f);
    y[row * 512 + tid]       = (x0 - mu) * rs * w[tid];
    y[row * 512 + tid + 256] = (x1 - mu) * rs * w[tid + 256];
}

// ---------------------------------------------------------------------------
// Causal depthwise conv (K=4) over xm = up[:, :1024], then SiLU -> xc
// ---------------------------------------------------------------------------
__global__ __launch_bounds__(256)
void k_conv_silu(const float* __restrict__ up, const float* __restrict__ cw,
                 const float* __restrict__ cb, float* __restrict__ xc)
{
    long long idx = (long long)blockIdx.x * 256 + threadIdx.x;  // < 8192*1024
    int c = (int)(idx & 1023);
    long long rowi = idx >> 10;       // b*512+s
    int s = (int)(rowi & 511);
    float acc = cb[c];
    #pragma unroll
    for (int j = 0; j < 4; ++j) {
        int sp = s - 3 + j;
        if (sp >= 0)
            acc += cw[j * 1024 + c] * up[(rowi - 3 + j) * 2048 + c];
    }
    xc[idx] = acc / (1.f + __expf(-acc));   // silu
}

// ---------------------------------------------------------------------------
// Gate projections: i_pre/f_pre[b,h,s] = concat(qh,kh,vh)[b,s,:] @ W + bias
// ---------------------------------------------------------------------------
__global__ __launch_bounds__(256)
void k_gates(const float* __restrict__ qh, const float* __restrict__ kh,
             const float* __restrict__ vh, const float* __restrict__ Wi,
             const float* __restrict__ bi, const float* __restrict__ Wf,
             const float* __restrict__ bf, float* __restrict__ ipre,
             float* __restrict__ fpre)
{
    __shared__ float red[8][256];
    const int row = blockIdx.x;       // b*S+s
    const int tid = threadIdx.x;
    float ai[4] = {0.f, 0.f, 0.f, 0.f};
    float af[4] = {0.f, 0.f, 0.f, 0.f};
    for (int k = tid; k < 3072; k += 256) {
        float x;
        if (k < 1024)      x = qh[(long long)row * 1024 + k];
        else if (k < 2048) x = kh[(long long)row * 1024 + k - 1024];
        else               x = vh[(long long)row * 1024 + k - 2048];
        const float* wi = Wi + (long long)k * 4;
        const float* wf = Wf + (long long)k * 4;
        #pragma unroll
        for (int hh = 0; hh < 4; ++hh) { ai[hh] += x * wi[hh]; af[hh] += x * wf[hh]; }
    }
    #pragma unroll
    for (int v = 0; v < 4; ++v) { red[v][tid] = ai[v]; red[4 + v][tid] = af[v]; }
    __syncthreads();
    for (int off = 128; off > 0; off >>= 1) {
        if (tid < off)
            for (int v = 0; v < 8; ++v) red[v][tid] += red[v][tid + off];
        __syncthreads();
    }
    if (tid < 4) {
        int b = row >> 9, s = row & 511;
        ipre[((long long)(b * 4 + tid)) * 512 + s] = red[tid][0] + bi[tid];
        fpre[((long long)(b * 4 + tid)) * 512 + s] = red[4 + tid][0] + bf[tid];
    }
}

// ---------------------------------------------------------------------------
// Per-(b,h) scan: F = cumsum(log_sigmoid(f_pre)), m_s = F_s + prefmax(i_t-F_t)
// ---------------------------------------------------------------------------
__global__ __launch_bounds__(512)
void k_scan(const float* __restrict__ ipre, const float* __restrict__ fpre,
            float* __restrict__ Fb, float* __restrict__ mb)
{
    __shared__ float sc[512];
    __shared__ float sm[512];
    const int bh = blockIdx.x, t = threadIdx.x;
    float f = fpre[(long long)bh * 512 + t];
    float lf = fminf(f, 0.f) - log1pf(__expf(-fabsf(f)));
    sc[t] = lf;
    __syncthreads();
    for (int off = 1; off < 512; off <<= 1) {
        float v = (t >= off) ? sc[t - off] : 0.f;
        __syncthreads();
        sc[t] += v;
        __syncthreads();
    }
    float F = sc[t];
    float g = ipre[(long long)bh * 512 + t] - F;
    sm[t] = g;
    __syncthreads();
    for (int off = 1; off < 512; off <<= 1) {
        float v = (t >= off) ? sm[t - off] : -1e30f;
        __syncthreads();
        sm[t] = fmaxf(sm[t], v);
        __syncthreads();
    }
    Fb[(long long)bh * 512 + t] = F;
    mb[(long long)bh * 512 + t] = F + sm[t];
}

// ---------------------------------------------------------------------------
// Fused mLSTM cell: per (b,h), 32-row stripe.
//   pass 1: C = qk*scale*exp(logD-m) via WMMA into LDS
//   reduce: norm = max(|rowsum C|, exp(-m))
//   pass 2: h = (C/norm) @ V via WMMA
// K/V tiles staged with GLOBAL_LOAD_ASYNC_TO_LDS (ASYNCcnt) when enabled.
// LDS ~151 KB of the 320 KB workgroup budget.
// ---------------------------------------------------------------------------
__global__ __launch_bounds__(256)
void k_cell(const float* __restrict__ qh, const float* __restrict__ kh,
            const float* __restrict__ vh, const float* __restrict__ Fb,
            const float* __restrict__ mb, const float* __restrict__ ipre,
            float* __restrict__ hcell)
{
    __shared__ float  Cbuf[32][512];      // 64 KB
    __shared__ __bf16 Qt[32][256];        // 16 KB
    __shared__ float  KVf[64][256];       // 64 KB, fp32 (V uses rows 0..31)
    __shared__ float  gS[512];
    __shared__ float  Fs[32], Ms[32], InvN[32];
    __shared__ float  red[32][8];

    const int tid  = threadIdx.x;
    const int lane = tid & 31;
    const int wid  = tid >> 5;
    const int lh   = lane >> 4;
    const int l16  = lane & 15;

    const int bh = blockIdx.y;            // b*NH + h
    const int b  = bh >> 2;
    const int hh = bh & 3;
    const int s0 = blockIdx.x * 32;

    for (int t = tid; t < 512; t += 256)
        gS[t] = ipre[(long long)bh * 512 + t] - Fb[(long long)bh * 512 + t];
    if (tid < 32) {
        Fs[tid] = Fb[(long long)bh * 512 + s0 + tid];
        Ms[tid] = mb[(long long)bh * 512 + s0 + tid];
    }
    {   // zero Cbuf (vectorized)
        float4 z4 = make_float4(0.f, 0.f, 0.f, 0.f);
        float4* c4 = (float4*)&Cbuf[0][0];
        for (int i = tid; i < 32 * 512 / 4; i += 256) c4[i] = z4;
    }
    {   // stage Q stripe 32x256 -> bf16 (vectorized)
        int r = tid >> 3, d0 = (tid & 7) * 32;
        const float4* src =
            (const float4*)(qh + ((long long)(b * 512 + s0 + r) * 4 + hh) * 256 + d0);
        #pragma unroll
        for (int i = 0; i < 4; ++i)
            bf8_from_f4x2(&Qt[r][d0 + i * 8], src[2 * i], src[2 * i + 1]);
    }
    __syncthreads();

    const float scale = 0.0625f;          // 1/sqrt(DH)
    const int tcmax = (s0 + 31) >> 6;     // inclusive last 64-wide t tile
    const int r = wid >> 2;               // 0..1 (row tile)
    const int c = wid & 3;                // 0..3 (col tile within 64)

#if USE_ASYNC_LDS
    const unsigned ldsKV = (unsigned)(unsigned long long)(&KVf[0][0]);
#endif

    // ---- pass 1: scores ----
    for (int tc = 0; tc <= tcmax; ++tc) {
        // stage K tile 64x256 fp32 into LDS
#if USE_ASYNC_LDS
        {
            #pragma unroll
            for (int it = 0; it < 16; ++it) {
                int chunk = it * 256 + tid;         // 16B chunk index, 0..4095
                int krow  = chunk >> 6;             // 64 chunks per 256-f row
                int dcol  = (chunk & 63) * 4;
                const float* gp =
                    kh + ((long long)(b * 512 + tc * 64 + krow) * 4 + hh) * 256 + dcol;
                unsigned laddr = ldsKV + (unsigned)chunk * 16u;
                asm volatile("global_load_async_to_lds_b128 %0, %1, off"
                             :: "v"(laddr), "v"(gp) : "memory");
            }
            asm volatile("s_wait_asynccnt 0x0" ::: "memory");
        }
#else
        {
            int kr = tid >> 2, d0 = (tid & 3) * 64;
            const float4* src =
                (const float4*)(kh + ((long long)(b * 512 + tc * 64 + kr) * 4 + hh) * 256 + d0);
            float4* dst = (float4*)&KVf[kr][d0];
            #pragma unroll
            for (int i = 0; i < 16; ++i) dst[i] = src[i];
        }
#endif
        __syncthreads();

        v8f acc = vzero8();
        #pragma unroll
        for (int ks = 0; ks < 8; ++ks) {
            v16bf a, bb;
            #pragma unroll
            for (int e = 0; e < 16; ++e) {
                int kk = ks * 32 + lh * 16 + e;
                a[e] = Qt[r * 16 + l16][kk];
            }
            #pragma unroll
            for (int e = 0; e < 16; ++e) {
                int kk = ks * 32 + lh * 16 + e;       // contiguous d
                bb[e] = (__bf16)KVf[c * 16 + l16][kk]; // B[d][t] = K[t][d]
            }
            acc = __builtin_amdgcn_wmma_f32_16x16x32_bf16(false, a, false, bb,
                                                          (short)0, acc, false, false);
        }
        #pragma unroll
        for (int vr = 0; vr < 8; ++vr) {
            int sl = r * 16 + vr + lh * 8;
            int tg = tc * 64 + c * 16 + l16;
            int sg = s0 + sl;
            float v = 0.f;
            if (tg <= sg)
                v = acc[vr] * scale * __expf(Fs[sl] - Ms[sl] + gS[tg]);
            Cbuf[sl][tc * 64 + c * 16 + l16] = v;
        }
        __syncthreads();
    }

    // ---- row sums -> norm ----
    {
        int rr = tid >> 3, seg = (tid & 7) * 64;
        float s = 0.f;
        for (int i = 0; i < 64; ++i) s += Cbuf[rr][seg + i];
        red[rr][tid & 7] = s;
    }
    __syncthreads();
    if (tid < 32) {
        float s = 0.f;
        for (int i = 0; i < 8; ++i) s += red[tid][i];
        float nrm = fmaxf(fabsf(s), __expf(-Ms[tid])) + 1e-6f;
        InvN[tid] = 1.f / nrm;
    }
    __syncthreads();

    // ---- pass 2: h = (C * invnorm) @ V ----
    v8f o00 = vzero8(), o01 = vzero8(), o10 = vzero8(), o11 = vzero8();
    const int nkt = blockIdx.x + 1;       // 32-wide t tiles up to causal edge
    for (int kt = 0; kt < nkt; ++kt) {
        // stage V tile 32x256 fp32 into LDS rows 0..31
#if USE_ASYNC_LDS
        {
            #pragma unroll
            for (int it = 0; it < 8; ++it) {
                int chunk = it * 256 + tid;         // 0..2047
                int vrow  = chunk >> 6;
                int dcol  = (chunk & 63) * 4;
                const float* gp =
                    vh + ((long long)(b * 512 + kt * 32 + vrow) * 4 + hh) * 256 + dcol;
                unsigned laddr = ldsKV + (unsigned)chunk * 16u;
                asm volatile("global_load_async_to_lds_b128 %0, %1, off"
                             :: "v"(laddr), "v"(gp) : "memory");
            }
            asm volatile("s_wait_asynccnt 0x0" ::: "memory");
        }
#else
        {
            int vr2 = tid >> 3, d0 = (tid & 7) * 32;
            const float4* src =
                (const float4*)(vh + ((long long)(b * 512 + kt * 32 + vr2) * 4 + hh) * 256 + d0);
            float4* dst = (float4*)&KVf[vr2][d0];
            #pragma unroll
            for (int i = 0; i < 8; ++i) dst[i] = src[i];
        }
#endif
        __syncthreads();
        #pragma unroll
        for (int r2 = 0; r2 < 2; ++r2) {
            #pragma unroll
            for (int ci = 0; ci < 2; ++ci) {
                int cc = wid + ci * 8;     // 0..15 col tiles over DH=256
                v16bf a, bb;
                #pragma unroll
                for (int e = 0; e < 16; ++e) {
                    int kk = lh * 16 + e;
                    int sl = r2 * 16 + l16;
                    a[e] = (__bf16)(Cbuf[sl][kt * 32 + kk] * InvN[sl]);
                }
                #pragma unroll
                for (int e = 0; e < 16; ++e)
                    bb[e] = (__bf16)KVf[lh * 16 + e][cc * 16 + l16];
                v8f acc = (r2 == 0) ? (ci == 0 ? o00 : o01) : (ci == 0 ? o10 : o11);
                acc = __builtin_amdgcn_wmma_f32_16x16x32_bf16(false, a, false, bb,
                                                              (short)0, acc, false, false);
                if (r2 == 0) { if (ci == 0) o00 = acc; else o01 = acc; }
                else         { if (ci == 0) o10 = acc; else o11 = acc; }
            }
        }
        __syncthreads();
    }

    // ---- store h (b,h,s,d) ----
    #pragma unroll
    for (int r2 = 0; r2 < 2; ++r2) {
        #pragma unroll
        for (int ci = 0; ci < 2; ++ci) {
            v8f acc = (r2 == 0) ? (ci == 0 ? o00 : o01) : (ci == 0 ? o10 : o11);
            int cc = wid + ci * 8;
            #pragma unroll
            for (int vr = 0; vr < 8; ++vr) {
                int sl = r2 * 16 + vr + lh * 8;
                int d = cc * 16 + l16;
                hcell[((long long)bh * 512 + s0 + sl) * 256 + d] = acc[vr];
            }
        }
    }
}

// ---------------------------------------------------------------------------
// h-norm over DH (no weight), * onorm, + skip*xc, * silu(z)  -> hst (B,S,INNER)
// ---------------------------------------------------------------------------
__global__ __launch_bounds__(256)
void k_hn_hst(const float* __restrict__ hcell, const float* __restrict__ xc,
              const float* __restrict__ up, const float* __restrict__ onorm,
              const float* __restrict__ skip, float* __restrict__ hst)
{
    __shared__ float r1[256], r2[256];
    const int tid = threadIdx.x;
    const int bh = blockIdx.x >> 9;
    const int s  = blockIdx.x & 511;
    const int b  = bh >> 2;
    const int hh = bh & 3;
    float x = hcell[(long long)blockIdx.x * 256 + tid];
    r1[tid] = x;
    r2[tid] = x * x;
    __syncthreads();
    for (int off = 128; off > 0; off >>= 1) {
        if (tid < off) { r1[tid] += r1[tid + off]; r2[tid] += r2[tid + off]; }
        __syncthreads();
    }
    float mu = r1[0] * (1.f / 256.f);
    float var = r2[0] * (1.f / 256.f) - mu * mu;
    float hn = (x - mu) * rsqrtf(var + 1e-5f);
    int col = hh * 256 + tid;
    long long row = (long long)b * 512 + s;
    hn *= onorm[col];
    float zz = up[row * 2048 + 1024 + col];
    float sz = zz / (1.f + __expf(-zz));
    hst[row * 1024 + col] = (hn + skip[col] * xc[row * 1024 + col]) * sz;
}

// ---------------------------------------------------------------------------
// Final: take last token, LN(post) then LN(out,w,b), then @ W_head + b_head
// ---------------------------------------------------------------------------
__global__ __launch_bounds__(256)
void k_head(const float* __restrict__ h, const float* __restrict__ postw,
            const float* __restrict__ ow, const float* __restrict__ ob,
            const float* __restrict__ Wh, const float* __restrict__ bhd,
            float* __restrict__ out)
{
    __shared__ float row[512];
    __shared__ float r1[256], r2[256];
    const int b = blockIdx.x, tid = threadIdx.x;
    const float* src = h + ((long long)(b * 512 + 511)) * 512;
    float x0 = src[tid], x1 = src[tid + 256];
    r1[tid] = x0 + x1;
    r2[tid] = x0 * x0 + x1 * x1;
    __syncthreads();
    for (int off = 128; off > 0; off >>= 1) {
        if (tid < off) { r1[tid] += r1[tid + off]; r2[tid] += r2[tid + off]; }
        __syncthreads();
    }
    float mu = r1[0] * (1.f / 512.f);
    float var = r2[0] * (1.f / 512.f) - mu * mu;
    float rs = rsqrtf(var + 1e-5f);
    float y0 = (x0 - mu) * rs * postw[tid];
    float y1 = (x1 - mu) * rs * postw[tid + 256];
    __syncthreads();
    r1[tid] = y0 + y1;
    r2[tid] = y0 * y0 + y1 * y1;
    __syncthreads();
    for (int off = 128; off > 0; off >>= 1) {
        if (tid < off) { r1[tid] += r1[tid + off]; r2[tid] += r2[tid + off]; }
        __syncthreads();
    }
    float mu2 = r1[0] * (1.f / 512.f);
    float var2 = r2[0] * (1.f / 512.f) - mu2 * mu2;
    float rs2 = rsqrtf(var2 + 1e-5f);
    row[tid]       = (y0 - mu2) * rs2 * ow[tid] + ob[tid];
    row[tid + 256] = (y1 - mu2) * rs2 * ow[tid + 256] + ob[tid + 256];
    __syncthreads();
    if (tid < 32) {
        float acc = bhd[tid];
        for (int k = 0; k < 512; ++k) acc += row[k] * Wh[k * 32 + tid];
        out[b * 32 + tid] = acc;
    }
}

// ---------------------------------------------------------------------------
extern "C" void kernel_launch(void* const* d_in, const int* in_sizes, int n_in,
                              void* d_out, int out_size, void* d_ws, size_t ws_size,
                              hipStream_t stream)
{
    (void)in_sizes; (void)n_in; (void)out_size; (void)ws_size;

    const float* inputs    = (const float*)d_in[0];
    const float* W_in      = (const float*)d_in[1];
    const float* ln_w      = (const float*)d_in[2];
    const float* W_up      = (const float*)d_in[3];
    const float* conv_w    = (const float*)d_in[4];
    const float* conv_b    = (const float*)d_in[5];
    const float* Wq        = (const float*)d_in[6];
    const float* Wk        = (const float*)d_in[7];
    const float* Wv        = (const float*)d_in[8];
    const float* W_i       = (const float*)d_in[9];
    const float* b_i       = (const float*)d_in[10];
    const float* W_f       = (const float*)d_in[11];
    const float* b_f       = (const float*)d_in[12];
    const float* skip      = (const float*)d_in[13];
    const float* onorm     = (const float*)d_in[14];
    const float* W_down    = (const float*)d_in[15];
    const float* post_ln_w = (const float*)d_in[16];
    const float* out_ln_w  = (const float*)d_in[17];
    const float* out_ln_b  = (const float*)d_in[18];
    const float* W_head    = (const float*)d_in[19];
    const float* b_head    = (const float*)d_in[20];

    float* ws   = (float*)d_ws;
    float* h    = ws;                       // 16*512*512
    float* xn   = h    + 4194304;           // 16*512*512
    float* up   = xn   + 4194304;           // 16*512*2048
    float* xc   = up   + 16777216;          // 16*512*1024
    float* qh   = xc   + 8388608;
    float* kh   = qh   + 8388608;
    float* vh   = kh   + 8388608;
    float* hc   = vh   + 8388608;
    float* ipre = hc   + 8388608;           // 16*4*512
    float* fpre = ipre + 32768;
    float* Fb   = fpre + 32768;
    float* mb   = Fb   + 32768;
    float* hst  = qh;                       // reuse qh after cell

    // h = inputs @ W_in   (M=8192, K=32, N=512)
    k_gemm_bf16<<<dim3(8, 128, 1), 256, 0, stream>>>(
        inputs, 32, 0, W_in, 512, 0, 0, h, 512, 0,
        nullptr, 0, 0, 8192, 512, 32);

    for (int bl = 0; bl < 2; ++bl) {
        // xn = LN(h)
        k_layernorm<<<8192, 256, 0, stream>>>(h, ln_w + bl * 512, xn);
        // up = xn @ W_up  (M=8192, K=512, N=2048)
        k_gemm_bf16<<<dim3(32, 128, 1), 256, 0, stream>>>(
            xn, 512, 0, W_up + (long long)bl * 512 * 2048, 2048, 0, 0,
            up, 2048, 0, nullptr, 0, 0, 8192, 2048, 512);
        // xc = silu(conv(xm) + b)
        k_conv_silu<<<32768, 256, 0, stream>>>(
            up, conv_w + bl * 4096, conv_b + bl * 1024, xc);
        // qh/kh from xc, vh from xm(=up), batched over NH with transposed W
        k_gemm_bf16<<<dim3(4, 128, 4), 256, 0, stream>>>(
            xc, 1024, 256, Wq + (long long)bl * 4 * 65536, 256, 65536, 1,
            qh, 1024, 256, nullptr, 0, 0, 8192, 256, 256);
        k_gemm_bf16<<<dim3(4, 128, 4), 256, 0, stream>>>(
            xc, 1024, 256, Wk + (long long)bl * 4 * 65536, 256, 65536, 1,
            kh, 1024, 256, nullptr, 0, 0, 8192, 256, 256);
        k_gemm_bf16<<<dim3(4, 128, 4), 256, 0, stream>>>(
            up, 2048, 256, Wv + (long long)bl * 4 * 65536, 256, 65536, 1,
            vh, 1024, 256, nullptr, 0, 0, 8192, 256, 256);
        // gates
        k_gates<<<8192, 256, 0, stream>>>(
            qh, kh, vh, W_i + (long long)bl * 3072 * 4, b_i + bl * 4,
            W_f + (long long)bl * 3072 * 4, b_f + bl * 4, ipre, fpre);
        // per-(b,h) scans: F and m
        k_scan<<<64, 512, 0, stream>>>(ipre, fpre, Fb, mb);
        // fused mLSTM cell
        k_cell<<<dim3(16, 64), 256, 0, stream>>>(qh, kh, vh, Fb, mb, ipre, hc);
        // normalization + gating -> hst (overwrites qh)
        k_hn_hst<<<32768, 256, 0, stream>>>(
            hc, xc, up, onorm + bl * 1024, skip + bl * 1024, hst);
        // h = h + hst @ W_down  (M=8192, K=1024, N=512), residual fused
        k_gemm_bf16<<<dim3(8, 128, 1), 256, 0, stream>>>(
            hst, 1024, 0, W_down + (long long)bl * 1024 * 512, 512, 0, 0,
            h, 512, 0, h, 512, 0, 8192, 512, 1024);
    }

    // final: last token, double LN, head
    k_head<<<16, 256, 0, stream>>>(h, post_ln_w, out_ln_w, out_ln_b,
                                   W_head, b_head, (float*)d_out);
}
